// AttnBlock_5669356834829
// MI455X (gfx1250) — matrix operand order repaired and checked by reading, not compiled, not run on hardware
//
#include <hip/hip_runtime.h>

// ---------------------------------------------------------------------------
// AttnBlock for MI455X (gfx1250, wave32, WMMA bf16)
// x:[2,512,64,64] fp32.  GroupNorm -> QKV(1x1) -> softmax(Q^T K /sqrt(c)) V
// -> proj(1x1) -> +x.  Flash-attention: 32 queries/WG, K/V tiles double-
// buffered in LDS via global_load_async_to_lds_b128 (ASYNCcnt-tracked),
// weights pre-swizzled to bf16 WMMA fragment layout.
// ---------------------------------------------------------------------------

constexpr int CH   = 512;    // channels
constexpr int HWD  = 4096;   // 64*64 pixels
constexpr int NB   = 2;      // batch
constexpr int CPG  = 16;     // channels per group (512/32)
constexpr float EPSV  = 1e-6f;
constexpr float SCALE = 0.04419417382415922f; // 512^-0.5

typedef __attribute__((ext_vector_type(16))) __bf16 v16bf;
typedef __attribute__((ext_vector_type(8)))  __bf16 v8bf;
typedef __attribute__((ext_vector_type(8)))  float  v8f;

__device__ __forceinline__ v16bf cat8(v8bf lo, v8bf hi) {
  return __builtin_shufflevector(lo, hi, 0,1,2,3,4,5,6,7,8,9,10,11,12,13,14,15);
}

#define WMMA_BF16(A, B, Cacc) \
  __builtin_amdgcn_wmma_f32_16x16x32_bf16(false, (A), false, (B), (short)0, (Cacc), false, false)
// reuse_a hint: caches A operand for the next (identical-opcode) WMMA
#define WMMA_BF16_RA(A, B, Cacc) \
  __builtin_amdgcn_wmma_f32_16x16x32_bf16(false, (A), false, (B), (short)0, (Cacc), true, false)

// CDNA5 async copy global->LDS (16B per lane), tracked by ASYNCcnt.
__device__ __forceinline__ void async_ld_b128(unsigned lds_off, const void* g) {
  asm volatile("global_load_async_to_lds_b128 %0, %1, off"
               :: "v"(lds_off), "v"(g) : "memory");
}
__device__ __forceinline__ void wait_async0() {
  asm volatile("s_wait_asynccnt 0" ::: "memory");
}

// ---------------------------------------------------------------------------
// Kernel 0: pre-swizzle the four 512x512 fp32 weight matrices into bf16 WMMA
// A-fragment order:  frag[(ot*16+ks)*32 + lane][e]  with contiguous K
// layout (lanes0-15: K = 32ks+0..15 ; lanes16-31: K = 32ks+16..31).
// ---------------------------------------------------------------------------
__global__ __launch_bounds__(256) void prep_w(
    const float* __restrict__ wq, const float* __restrict__ wk,
    const float* __restrict__ wv, const float* __restrict__ wo,
    __bf16* __restrict__ fq, __bf16* __restrict__ fk,
    __bf16* __restrict__ fv, __bf16* __restrict__ fo)
{
  int tid = blockIdx.x * 256 + threadIdx.x;     // 0..65535
  int m   = tid >> 14;                          // which matrix
  int r   = tid & 16383;                        // (ot*16+ks)*32 + lane
  int lane = r & 31;
  int otks = r >> 5;
  int ks   = otks & 15;
  int ot   = otks >> 4;
  const float* W = (m == 0) ? wq : (m == 1) ? wk : (m == 2) ? wv : wo;
  __bf16*      F = (m == 0) ? fq : (m == 1) ? fk : (m == 2) ? fv : fo;
  int row = ot * 16 + (lane & 15);
  int kb  = (lane < 16) ? 0 : 16;
  const float* src = W + (size_t)row * CH + ks * 32 + kb;
  v16bf o;
  #pragma unroll
  for (int e = 0; e < 8; e++) {
    o[e]     = (__bf16)src[e];
    o[8 + e] = (__bf16)src[8 + e];
  }
  *(v16bf*)(F + (size_t)r * 16) = o;
}

// ---------------------------------------------------------------------------
// Kernel 1: GroupNorm -> hn (bf16, pixel-major [b][p][c]).
// ---------------------------------------------------------------------------
__global__ __launch_bounds__(256) void gn_kernel(
    const float* __restrict__ x, const float* __restrict__ gamma,
    const float* __restrict__ beta, __bf16* __restrict__ hn)
{
  const int grp = blockIdx.x;        // 0..63
  const int b   = grp >> 5;
  const int g   = grp & 31;
  const int c0  = g * CPG;
  const int t   = threadIdx.x;
  const float* xb = x + ((size_t)b * CH + c0) * HWD;   // 65536 contiguous floats

  float s = 0.f, s2 = 0.f;
  for (int i = t; i < CPG * HWD; i += 256) {
    float v = xb[i];
    s += v; s2 += v * v;
  }
  __shared__ float rs[256], rq[256];
  rs[t] = s; rq[t] = s2;
  __syncthreads();
  for (int o = 128; o > 0; o >>= 1) {
    if (t < o) { rs[t] += rs[t + o]; rq[t] += rq[t + o]; }
    __syncthreads();
  }
  const float inv_n = 1.f / (CPG * HWD);
  const float mean = rs[0] * inv_n;
  const float var  = rq[0] * inv_n - mean * mean;
  const float rstd = rsqrtf(var + EPSV);

  float a[CPG], sh[CPG];
  #pragma unroll
  for (int ch = 0; ch < CPG; ch++) {
    float gm = gamma[c0 + ch];
    a[ch]  = gm * rstd;
    sh[ch] = beta[c0 + ch] - mean * gm * rstd;
  }
  for (int p = t; p < HWD; p += 256) {
    v16bf o;
    #pragma unroll
    for (int ch = 0; ch < CPG; ch++)
      o[ch] = (__bf16)(xb[(size_t)ch * HWD + p] * a[ch] + sh[ch]);
    *(v16bf*)(hn + ((size_t)b * HWD + p) * CH + c0) = o;   // 32B coalesced
  }
}

// ---------------------------------------------------------------------------
// Kernel 2: QKV projections; pure b128-load + wmma inner loop.
// ---------------------------------------------------------------------------
__global__ __launch_bounds__(256) void qkv_kernel(
    const __bf16* __restrict__ hn,
    const __bf16* __restrict__ fq, const float* __restrict__ bq,
    const __bf16* __restrict__ fk, const float* __restrict__ bk,
    const __bf16* __restrict__ fv, const float* __restrict__ bv,
    __bf16* __restrict__ qt, __bf16* __restrict__ kt, __bf16* __restrict__ vc)
{
  const int wid  = blockIdx.x * 8 + (threadIdx.x >> 5);
  const int pt   = wid & 255;
  const int rest = wid >> 8;
  const int ot   = rest & 31;
  const int b    = rest >> 5;
  const int p0 = pt * 16, o0 = ot * 16;
  const int lane = threadIdx.x & 31, hf = lane >> 4, l15 = lane & 15;

  v8f cq = {}, ck = {}, cv = {};
  const __bf16* brow  = hn + ((size_t)b * HWD + p0 + l15) * CH;
  const size_t  fbase = ((size_t)ot * 16 * 32 + lane) * 16;   // + ks*32*16

  for (int ks = 0; ks < 16; ks++) {
    v16bf Bf = *(const v16bf*)(brow + ks * 32 + hf * 16);
    v16bf Aq = *(const v16bf*)(fq + fbase + (size_t)ks * 512);
    v16bf Ak = *(const v16bf*)(fk + fbase + (size_t)ks * 512);
    v16bf Av = *(const v16bf*)(fv + fbase + (size_t)ks * 512);
    cq = WMMA_BF16(Aq, Bf, cq);
    ck = WMMA_BF16(Ak, Bf, ck);
    cv = WMMA_BF16(Av, Bf, cv);
  }
  #pragma unroll
  for (int r = 0; r < 8; r++) {
    int o = o0 + r + hf * 8;
    int p = p0 + l15;
    qt[((size_t)b * HWD + p) * CH + o] = (__bf16)((cq[r] + bq[o]) * SCALE);
    kt[((size_t)b * HWD + p) * CH + o] = (__bf16)(ck[r] + bk[o]);
    vc[((size_t)b * CH + o) * HWD + p] = (__bf16)(cv[r] + bv[o]);
  }
}

// ---------------------------------------------------------------------------
// Kernel 3: flash attention, 32 queries per workgroup.
// 8 waves = 2 query-subtiles x 4 channel-slices (128 ch each).
// K tile [32j][512c] and V tile [512c][32j] double-buffered in dynamic LDS,
// filled with global_load_async_to_lds_b128; copy of tile i+1 overlaps
// compute of tile i.  grid = NB*(HWD/32) = 256, block = 256,
// dynamic LDS = 2*32K (K) + 2*32K (V) + 16K (S scratch) = 144 KB.
// ---------------------------------------------------------------------------
__global__ __launch_bounds__(256) void flash_kernel(
    const __bf16* __restrict__ qt, const __bf16* __restrict__ kt,
    const __bf16* __restrict__ vc, __bf16* __restrict__ at)
{
  extern __shared__ char smem[];
  __bf16* sK    = (__bf16*)smem;                    // [2][32*512]
  __bf16* sV    = (__bf16*)(smem + 2 * 32768);      // [2][512*32]
  float*  sPart = (float*) (smem + 4 * 32768);      // [4 cc][32 q][32 j]

  const int b  = blockIdx.x >> 7;            // HWD/32 = 128 query tiles
  const int q0 = (blockIdx.x & 127) * 32;
  const int w    = threadIdx.x >> 5;
  const int lane = threadIdx.x & 31, hf = lane >> 4, l15 = lane & 15;
  const int cc = w & 3, qs = w >> 2;         // channel slice, query subtile
  const int cslice = cc * 128;
  const int row = l15;                       // lane's softmax row (dup halves)
  const int kb  = hf * 16;                   // A-fragment K-offset pattern

  // ---- stage one 32-key K/V tile into LDS buffer `buf` (async) ----
  const int t = threadIdx.x;
  auto stage = [&](int buf, int jb) {
    // K: flat [32 rows][512 ch]; chunk m = t + 256*s; LDS & global step match
    const __bf16* kg = kt + ((size_t)b * HWD + jb) * CH
                          + (size_t)(t >> 6) * CH + (t & 63) * 8;
    unsigned lk = (unsigned)(size_t)(sK + buf * 16384) + t * 16;
    #pragma unroll
    for (int s = 0; s < 8; s++)
      async_ld_b128(lk + s * 4096, (const char*)kg + s * 4096);
    // V: LDS [512 ch][32 j]; chunk m: ch = m/4 (+64/s), off = (m%4)*8
    const __bf16* vg = vc + (size_t)b * CH * HWD + jb
                          + (size_t)(t >> 2) * HWD + (t & 3) * 8;
    unsigned lv = (unsigned)(size_t)(sV + buf * 16384) + t * 16;
    #pragma unroll
    for (int s = 0; s < 8; s++)
      async_ld_b128(lv + s * 4096, vg + (size_t)64 * s * HWD);
  };

  // ---- Q A-fragments: 4 x 32-channel slices, register-resident ----
  const __bf16* qrow = qt + ((size_t)b * HWD + q0 + qs * 16 + l15) * CH + cslice;
  v16bf aQ[4];
  #pragma unroll
  for (int f = 0; f < 4; f++)
    aQ[f] = cat8(*(const v8bf*)(qrow + 32 * f + hf * 16),
                 *(const v8bf*)(qrow + 32 * f + 8 + hf * 16));

  v8f O[8] = {};                              // 16q x 128c accumulator
  float m_i = -__builtin_inff(), l_i = 0.f;

  stage(0, 0);                                // prologue: tile 0 in flight

  for (int it = 0; it < HWD / 32; ++it) {
    const int cur = it & 1;
    wait_async0();                            // own-wave tile `it` loads done
    __syncthreads();                          // all waves staged; prev buffer free
    if (it + 1 < HWD / 32) stage(cur ^ 1, (it + 1) * 32);

    // ---- partial S over 128 channels (two 16-col subtiles) from LDS K ----
    const __bf16* k0 = sK + cur * 16384 + (size_t)l15 * 512 + cslice + hf * 16;
    v8f S0 = {}, S1 = {};
    #pragma unroll
    for (int f = 0; f < 4; f++) {
      v16bf B0 = *(const v16bf*)(k0 + 32 * f);
      v16bf B1 = *(const v16bf*)(k0 + 16 * 512 + 32 * f);
      S0 = WMMA_BF16(aQ[f], B0, S0);
      S1 = WMMA_BF16(aQ[f], B1, S1);
    }
    #pragma unroll
    for (int r = 0; r < 8; r++) {
      int qq = qs * 16 + r + hf * 8;
      sPart[(cc * 32 + qq) * 32 + l15]      = S0[r];
      sPart[(cc * 32 + qq) * 32 + 16 + l15] = S1[r];
    }
    __syncthreads();

    // ---- softmax: fold 4-slice reduction into the LDS reads ----
    const float* pr = sPart + (qs * 16 + row) * 32;
    float4 a0 = {0,0,0,0}, a1 = {0,0,0,0}, a2 = {0,0,0,0}, a3 = {0,0,0,0};
    #pragma unroll
    for (int c4 = 0; c4 < 4; c4++) {
      const float* p4 = pr + c4 * 1024;
      float4 u0 = *(const float4*)(p4 + kb);
      float4 u1 = *(const float4*)(p4 + kb + 4);
      float4 u2 = *(const float4*)(p4 + kb + 8);
      float4 u3 = *(const float4*)(p4 + kb + 12);
      a0.x+=u0.x; a0.y+=u0.y; a0.z+=u0.z; a0.w+=u0.w;
      a1.x+=u1.x; a1.y+=u1.y; a1.z+=u1.z; a1.w+=u1.w;
      a2.x+=u2.x; a2.y+=u2.y; a2.z+=u2.z; a2.w+=u2.w;
      a3.x+=u3.x; a3.y+=u3.y; a3.z+=u3.z; a3.w+=u3.w;
    }
    float sv[16] = {a0.x,a0.y,a0.z,a0.w, a1.x,a1.y,a1.z,a1.w,
                    a2.x,a2.y,a2.z,a2.w, a3.x,a3.y,a3.z,a3.w};
    float mloc = -__builtin_inff();
    #pragma unroll
    for (int e = 0; e < 16; e++) mloc = fmaxf(mloc, sv[e]);
    mloc = fmaxf(mloc, __shfl_xor(mloc, 16));   // union of halves = full row
    float mnew  = fmaxf(m_i, mloc);
    float alpha = __expf(m_i - mnew);
    float pe[16];
    float lloc = 0.f;
    #pragma unroll
    for (int e = 0; e < 16; e++) { pe[e] = __expf(sv[e] - mnew); lloc += pe[e]; }
    lloc += __shfl_xor(lloc, 16);
    l_i = l_i * alpha + lloc;
    m_i = mnew;

    #pragma unroll
    for (int r = 0; r < 8; r++) {          // rescale O rows by alpha(row)
      float ar = __shfl(alpha, r + hf * 8);
      #pragma unroll
      for (int u = 0; u < 8; u++) O[u][r] *= ar;
    }

    v16bf aP;                               // P fragment (contiguous K: j = 16*hf + e)
    #pragma unroll
    for (int e = 0; e < 16; e++) aP[e] = (__bf16)pe[e];

    // ---- O += P * V^T over this wave's 8 channel subtiles (A reused) ----
    const __bf16* vbase = sV + cur * 16384 + (size_t)(cslice + l15) * 32 + hf * 16;
    #pragma unroll
    for (int u = 0; u < 8; u++) {
      v16bf Bv = *(const v16bf*)(vbase + u * 512);   // +16 channels = 16*32 el
      O[u] = (u < 7) ? WMMA_BF16_RA(aP, Bv, O[u]) : WMMA_BF16(aP, Bv, O[u]);
    }
  }

  // ---- normalize by row sum, write attn out pixel-major bf16 ----
  const float rinv = 1.f / l_i;
  #pragma unroll
  for (int r = 0; r < 8; r++) {
    float rv = __shfl(rinv, r + hf * 8);
    int qq = q0 + qs * 16 + r + hf * 8;
    size_t base = ((size_t)b * HWD + qq) * CH + cslice + l15;
    #pragma unroll
    for (int u = 0; u < 8; u++)
      at[base + u * 16] = (__bf16)(O[u][r] * rv);
  }
}

// ---------------------------------------------------------------------------
// Kernel 4: output projection + bias + residual (fp32 channel-major out).
// ---------------------------------------------------------------------------
__global__ __launch_bounds__(256) void proj_kernel(
    const __bf16* __restrict__ at, const __bf16* __restrict__ fo,
    const float* __restrict__ bo, const float* __restrict__ x,
    float* __restrict__ out)
{
  const int wid  = blockIdx.x * 8 + (threadIdx.x >> 5);
  const int pt   = wid & 255;
  const int rest = wid >> 8;
  const int ot   = rest & 31;
  const int b    = rest >> 5;
  const int p0 = pt * 16, o0 = ot * 16;
  const int lane = threadIdx.x & 31, hf = lane >> 4, l15 = lane & 15;

  v8f acc = {};
  const __bf16* brow  = at + ((size_t)b * HWD + p0 + l15) * CH;
  const size_t  fbase = ((size_t)ot * 16 * 32 + lane) * 16;

  for (int ks = 0; ks < 16; ks++) {
    v16bf Bf = *(const v16bf*)(brow + ks * 32 + hf * 16);
    v16bf A  = *(const v16bf*)(fo + fbase + (size_t)ks * 512);
    acc = WMMA_BF16(A, Bf, acc);
  }
  #pragma unroll
  for (int r8 = 0; r8 < 8; r8++) {
    int o = o0 + r8 + hf * 8;
    size_t idx = ((size_t)b * CH + o) * HWD + p0 + l15;
    out[idx] = x[idx] + acc[r8] + bo[o];
  }
}

// ---------------------------------------------------------------------------
extern "C" void kernel_launch(void* const* d_in, const int* in_sizes, int n_in,
                              void* d_out, int out_size, void* d_ws, size_t ws_size,
                              hipStream_t stream) {
  (void)in_sizes; (void)n_in; (void)out_size; (void)ws_size;
  const float* x     = (const float*)d_in[0];
  const float* gamma = (const float*)d_in[1];
  const float* beta  = (const float*)d_in[2];
  const float* wq    = (const float*)d_in[3];
  const float* bq    = (const float*)d_in[4];
  const float* wk    = (const float*)d_in[5];
  const float* bk    = (const float*)d_in[6];
  const float* wv    = (const float*)d_in[7];
  const float* bv    = (const float*)d_in[8];
  const float* wo    = (const float*)d_in[9];
  const float* bo    = (const float*)d_in[10];

  char* ws = (char*)d_ws;
  const size_t BUF  = (size_t)NB * HWD * CH * sizeof(__bf16);  // 8 MiB
  const size_t WBUF = (size_t)CH * CH * sizeof(__bf16);        // 512 KiB
  __bf16* hn = (__bf16*)(ws);
  __bf16* qt = (__bf16*)(ws + 1 * BUF);
  __bf16* kt = (__bf16*)(ws + 2 * BUF);
  __bf16* vc = (__bf16*)(ws + 3 * BUF);
  __bf16* at = (__bf16*)(ws + 4 * BUF);
  __bf16* fq = (__bf16*)(ws + 5 * BUF);
  __bf16* fk = (__bf16*)(ws + 5 * BUF + 1 * WBUF);
  __bf16* fv = (__bf16*)(ws + 5 * BUF + 2 * WBUF);
  __bf16* fo = (__bf16*)(ws + 5 * BUF + 3 * WBUF);

  prep_w      <<<256, 256, 0, stream>>>(wq, wk, wv, wo, fq, fk, fv, fo);
  gn_kernel   <<<NB * 32, 256, 0, stream>>>(x, gamma, beta, hn);
  qkv_kernel  <<<NB * 32 * 256 / 8, 256, 0, stream>>>(hn, fq, bq, fk, bk, fv, bv, qt, kt, vc);
  flash_kernel<<<NB * (HWD / 32), 256, 144 * 1024, stream>>>(qt, kt, vc, at);
  proj_kernel <<<NB * 32 * 256 / 8, 256, 0, stream>>>(at, fo, bo, x, (float*)d_out);
}
